// SRModel_3023656976557
// MI455X (gfx1250) — matrix operand order repaired and checked by reading, not compiled, not run on hardware
//
#include <hip/hip_runtime.h>
#include <math.h>

typedef __attribute__((ext_vector_type(16))) _Float16 v16h;
typedef __attribute__((ext_vector_type(8)))  _Float16 v8h;
typedef __attribute__((ext_vector_type(8)))  float    v8f;

#define DEV __device__ __forceinline__

DEV float gelu_exact(float x) { return 0.5f * x * (1.0f + erff(x * 0.70710678118654752f)); }

// ---------------------------------------------------------------------------
// Weight pack: fp32 [K][N] -> f16 transposed+padded [Np][Kp]
// ---------------------------------------------------------------------------
__global__ void pack_wT(const float* __restrict__ w, _Float16* __restrict__ o,
                        int K, int N, int Kp, int Np) {
    int id = blockIdx.x * blockDim.x + threadIdx.x;
    if (id >= Kp * Np) return;
    int n = id / Kp, k = id % Kp;
    float v = (k < K && n < N) ? w[k * N + n] : 0.0f;
    o[(size_t)n * Kp + k] = (_Float16)v;
}

// ---------------------------------------------------------------------------
// Bicubic 4x upsample (Keys a=-0.5, border-renormalized), NCHW 64x64 -> 256x256
// ---------------------------------------------------------------------------
DEV float cub_w(float x) {
    x = fabsf(x);
    if (x <= 1.0f) return ((1.5f * x - 2.5f) * x) * x + 1.0f;
    if (x < 2.0f)  return ((-0.5f * x + 2.5f) * x - 4.0f) * x + 2.0f;
    return 0.0f;
}

__global__ void bicubic_up(const float* __restrict__ lr, float* __restrict__ up) {
    int id = blockIdx.x * blockDim.x + threadIdx.x;
    if (id >= 2 * 3 * 256 * 256) return;
    int x = id & 255;
    int y = (id >> 8) & 255;
    int t = id >> 16;
    int c = t % 3, b = t / 3;
    float sx = (x + 0.5f) * 0.25f - 0.5f;
    float sy = (y + 0.5f) * 0.25f - 0.5f;
    int ix0 = (int)floorf(sx), iy0 = (int)floorf(sy);
    float fx = sx - ix0, fy = sy - iy0;
    float wx[4], wy[4];
    int   jx[4], jy[4];
    float swx = 0.0f, swy = 0.0f;
    #pragma unroll
    for (int i = 0; i < 4; ++i) {
        int tx = ix0 - 1 + i, ty = iy0 - 1 + i;
        float wxx = cub_w(fx - (float)(i - 1));
        float wyy = cub_w(fy - (float)(i - 1));
        if (tx < 0 || tx > 63) wxx = 0.0f;
        if (ty < 0 || ty > 63) wyy = 0.0f;
        jx[i] = min(max(tx, 0), 63); jy[i] = min(max(ty, 0), 63);
        wx[i] = wxx; wy[i] = wyy; swx += wxx; swy += wyy;
    }
    float inv = 1.0f / (swx * swy);
    const float* src = lr + (((size_t)b * 3 + c) << 12);
    float acc = 0.0f;
    #pragma unroll
    for (int i = 0; i < 4; ++i)
        #pragma unroll
        for (int j = 0; j < 4; ++j)
            acc += wy[i] * wx[j] * src[(jy[i] << 6) + jx[j]];
    up[(((size_t)b * 3 + c) << 16) + (y << 8) + x] = acc * inv;
}

// ---------------------------------------------------------------------------
// Encoder convs (3x3 SAME, zero pad, GELU), fp32 direct
// ---------------------------------------------------------------------------
__global__ void enc_conv0(const float* __restrict__ in, const float* __restrict__ w,
                          const float* __restrict__ bias, float* __restrict__ out) {
    int id = blockIdx.x * blockDim.x + threadIdx.x;
    if (id >= 2 * 64 * 64 * 64) return;
    int x = id & 63, y = (id >> 6) & 63, t = id >> 12;
    int co = t & 63, b = t >> 6;
    float acc = bias[co];
    #pragma unroll
    for (int ci = 0; ci < 3; ++ci) {
        const float* ip = in + (((size_t)b * 3 + ci) << 12);
        const float* wp = w + (co * 3 + ci) * 9;
        #pragma unroll
        for (int ky = 0; ky < 3; ++ky) {
            int iy = y + ky - 1;
            if (iy < 0 || iy > 63) continue;
            #pragma unroll
            for (int kx = 0; kx < 3; ++kx) {
                int ix = x + kx - 1;
                if (ix < 0 || ix > 63) continue;
                acc += wp[ky * 3 + kx] * ip[(iy << 6) + ix];
            }
        }
    }
    out[(((size_t)b * 64 + co) << 12) + (y << 6) + x] = gelu_exact(acc);
}

__global__ void enc_convh(const float* __restrict__ in, const float* __restrict__ w,
                          const float* __restrict__ bias, float* __restrict__ out) {
    int id = blockIdx.x * blockDim.x + threadIdx.x;
    if (id >= 2 * 64 * 64 * 64) return;
    int x = id & 63, y = (id >> 6) & 63, t = id >> 12;
    int co = t & 63, b = t >> 6;
    float acc = bias[co];
    for (int ci = 0; ci < 64; ++ci) {
        const float* ip = in + (((size_t)b * 64 + ci) << 12);
        const float* wp = w + (co * 64 + ci) * 9;
        #pragma unroll
        for (int ky = 0; ky < 3; ++ky) {
            int iy = y + ky - 1;
            if (iy < 0 || iy > 63) continue;
            #pragma unroll
            for (int kx = 0; kx < 3; ++kx) {
                int ix = x + kx - 1;
                if (ix < 0 || ix > 63) continue;
                acc += wp[ky * 3 + kx] * ip[(iy << 6) + ix];
            }
        }
    }
    out[(((size_t)b * 64 + co) << 12) + (y << 6) + x] = gelu_exact(acc);
}

// ---------------------------------------------------------------------------
// Build MLP input rows: [64 gathered feats | 40 PE | fx fy | 22 zero pad] f16
// ---------------------------------------------------------------------------
__global__ void build_X(const float* __restrict__ feat, _Float16* __restrict__ X) {
    int row = blockIdx.x * blockDim.x + threadIdx.x;
    if (row >= 131072) return;
    int b = row >> 16;
    int p = row & 65535;
    int hy = p >> 8, hx = p & 255;
    float lx = (hx + 0.5f) * 0.25f - 0.5f;
    float ly = (hy + 0.5f) * 0.25f - 0.5f;
    float cxf = fminf(fmaxf(rintf(lx), 0.0f), 63.0f);
    float cyf = fminf(fmaxf(rintf(ly), 0.0f), 63.0f);
    int cx = (int)cxf, cy = (int)cyf;
    _Float16* xr = X + (size_t)row * 128;
    const float* fb = feat + ((size_t)b << 18) + ((size_t)cy << 6) + cx;
    #pragma unroll 8
    for (int c = 0; c < 64; ++c) xr[c] = (_Float16)fb[(size_t)c << 12];
    float xn = (hx + 0.5f) * (2.0f / 256.0f) - 1.0f;
    float yn = (hy + 0.5f) * (2.0f / 256.0f) - 1.0f;
    float fr = 3.14159265358979323846f;
    #pragma unroll
    for (int j = 0; j < 10; ++j) {
        float px = xn * fr, py = yn * fr;
        xr[64 + 4 * j + 0] = (_Float16)sinf(px);
        xr[64 + 4 * j + 1] = (_Float16)cosf(px);
        xr[64 + 4 * j + 2] = (_Float16)sinf(py);
        xr[64 + 4 * j + 3] = (_Float16)cosf(py);
        fr *= 2.0f;
    }
    xr[104] = (_Float16)(lx - cxf);
    xr[105] = (_Float16)(ly - cyf);
    #pragma unroll
    for (int k = 106; k < 128; ++k) xr[k] = (_Float16)0.0f;
}

// ---------------------------------------------------------------------------
// WMMA GEMM v2: C[M,N] = act(A[M,Kp] f16 @ Bt[N,Kp]^T f16 + bias)
//
// Block = 8 waves, block tile = 512 rows x 64 cols.
// The 64xKp f16 weight slab (contiguous in Bt) is staged once into LDS and
// shared by all 8 waves. Each wave owns 4 M-tiles x 4 N-tiles (16 v8f accs);
// per 32-K step: 8 global b128 (A) + 4 LDS 32B reads (B) feed 16 WMMAs.
//
// A-frag: lane m=lane&15, 16 contiguous K at k0 + 16*(lane>>4) (same order as B).
// B-frag: lane n=lane&15, 16 contiguous K at k0 + 16*(lane>>4).
// All tile counts divide exactly -> no predication; EXEC all-ones at WMMA.
// ---------------------------------------------------------------------------
template <int KP, bool GELU_ACT, bool F32OUT>
__global__ __launch_bounds__(256) void gemm_wmma(
        const _Float16* __restrict__ A,
        const _Float16* __restrict__ Bt,
        const float* __restrict__ bias, int nbias,
        _Float16* __restrict__ C16, float* __restrict__ C32,
        int M, int N) {
    __shared__ _Float16 Bs[64 * KP];

    int nslab = N >> 6;
    int slab  = blockIdx.x % nslab;
    int mblk  = blockIdx.x / nslab;
    int col0  = slab << 6;
    int rowB  = mblk << 9;             // 512 rows per block

    // Cooperative stage of weight slab [64][KP] (contiguous region of Bt)
    {
        const v8h* src = (const v8h*)(Bt + (size_t)col0 * KP);
        v8h* dst = (v8h*)Bs;
        #pragma unroll
        for (int i = threadIdx.x; i < (64 * KP) / 8; i += 256)
            dst[i] = src[i];
    }
    __syncthreads();

    int lane = threadIdx.x & 31;
    int wave = threadIdx.x >> 5;
    int row0 = rowB + (wave << 6);     // 4 M-tiles of 16 rows per wave
    int aoff = (lane >> 4) << 4;       // 0 or 16
    int boff = (lane >> 4) << 4;       // 0 or 16
    int lanN = lane & 15;

    const _Float16* arow[4];
    #pragma unroll
    for (int mt = 0; mt < 4; ++mt)
        arow[mt] = A + (size_t)(row0 + (mt << 4) + lanN) * KP + aoff;

    const v8f vz = {0.f, 0.f, 0.f, 0.f, 0.f, 0.f, 0.f, 0.f};
    v8f acc[4][4];
    #pragma unroll
    for (int mt = 0; mt < 4; ++mt)
        #pragma unroll
        for (int t = 0; t < 4; ++t) acc[mt][t] = vz;

    #pragma unroll
    for (int k0 = 0; k0 < KP; k0 += 32) {
        union { v16h v; v8h h[2]; } a[4];
        #pragma unroll
        for (int mt = 0; mt < 4; ++mt) {
            a[mt].h[0] = *(const v8h*)(arow[mt] + k0);
            a[mt].h[1] = *(const v8h*)(arow[mt] + k0 + 8);
        }
        #pragma unroll
        for (int t = 0; t < 4; ++t) {
            v16h bf = *(const v16h*)(Bs + (size_t)((t << 4) + lanN) * KP + k0 + boff);
            #pragma unroll
            for (int mt = 0; mt < 4; ++mt)
                acc[mt][t] = __builtin_amdgcn_wmma_f32_16x16x32_f16(
                    false, a[mt].v, false, bf, (short)0, acc[mt][t], false, false);
        }
    }

    int mloc = (lane >> 4) << 3;       // D: VGPR j holds M=j (+8 for hi lanes)
    #pragma unroll
    for (int t = 0; t < 4; ++t) {
        int c = col0 + (t << 4) + lanN;
        float bv = (c < nbias) ? bias[c] : 0.0f;
        #pragma unroll
        for (int mt = 0; mt < 4; ++mt) {
            #pragma unroll
            for (int j = 0; j < 8; ++j) {
                int r = row0 + (mt << 4) + mloc + j;
                float v = acc[mt][t][j] + bv;
                if (GELU_ACT) v = gelu_exact(v);
                if (F32OUT) C32[(size_t)r * N + c] = v;
                else        C16[(size_t)r * N + c] = (_Float16)v;
            }
        }
    }
}

// ---------------------------------------------------------------------------
// Softmax over 49 logits + 7x7 weighted patch blend -> xcat = [lr_up | res0]
// ---------------------------------------------------------------------------
__global__ void softmax_patch(const float* __restrict__ logits,
                              const float* __restrict__ lr,
                              const float* __restrict__ lr_up,
                              float* __restrict__ xcat) {
    int row = blockIdx.x * blockDim.x + threadIdx.x;
    if (row >= 131072) return;
    int b = row >> 16;
    int p = row & 65535;
    int hy = p >> 8, hx = p & 255;
    int cx = (int)fminf(fmaxf(rintf((hx + 0.5f) * 0.25f - 0.5f), 0.0f), 63.0f);
    int cy = (int)fminf(fmaxf(rintf((hy + 0.5f) * 0.25f - 0.5f), 0.0f), 63.0f);

    const float* lg = logits + ((size_t)row << 6);
    float mx = lg[0];
    #pragma unroll
    for (int k = 1; k < 49; ++k) mx = fmaxf(mx, lg[k]);
    float w[49];
    float s = 0.0f;
    #pragma unroll
    for (int k = 0; k < 49; ++k) { w[k] = expf(lg[k] - mx); s += w[k]; }
    float inv = 1.0f / s;

    #pragma unroll
    for (int c = 0; c < 3; ++c) {
        const float* ip = lr + (((size_t)b * 3 + c) << 12);
        float acc = 0.0f;
        int k = 0;
        #pragma unroll
        for (int oy = -3; oy <= 3; ++oy) {
            int ny = min(max(cy + oy, 0), 63);
            #pragma unroll
            for (int ox = -3; ox <= 3; ++ox) {
                int nx = min(max(cx + ox, 0), 63);
                acc += w[k] * ip[(ny << 6) + nx];
                ++k;
            }
        }
        float sr = acc * inv;
        float lu = lr_up[(((size_t)b * 3 + c) << 16) + p];
        xcat[(((size_t)b * 6 + c) << 16) + p]     = lu;
        xcat[(((size_t)b * 6 + 3 + c) << 16) + p] = sr - lu;
    }
}

// ---------------------------------------------------------------------------
// Grouped 3x3 conv (groups=3, SAME zero pad), optional GELU
// ---------------------------------------------------------------------------
__global__ void gconv3x3(const float* __restrict__ in, const float* __restrict__ w,
                         const float* __restrict__ bias, float* __restrict__ out,
                         int Cin, int Cout, int act) {
    int id = blockIdx.x * blockDim.x + threadIdx.x;
    if (id >= 2 * Cout * 65536) return;
    int x = id & 255, y = (id >> 8) & 255, t = id >> 16;
    int co = t % Cout, b = t / Cout;
    int cing = Cin / 3, coutg = Cout / 3;
    int g = co / coutg;
    float acc = bias[co];
    for (int ci = 0; ci < cing; ++ci) {
        const float* ip = in + (((size_t)b * Cin + g * cing + ci) << 16);
        const float* wp = w + (co * cing + ci) * 9;
        #pragma unroll
        for (int ky = 0; ky < 3; ++ky) {
            int iy = y + ky - 1;
            if (iy < 0 || iy > 255) continue;
            #pragma unroll
            for (int kx = 0; kx < 3; ++kx) {
                int ix = x + kx - 1;
                if (ix < 0 || ix > 255) continue;
                acc += wp[ky * 3 + kx] * ip[(iy << 8) + ix];
            }
        }
    }
    if (act) acc = gelu_exact(acc);
    out[(((size_t)b * Cout + co) << 16) + (y << 8) + x] = acc;
}

// ---------------------------------------------------------------------------
// Final: sr_pred = lr_up+res0+rr; out = clip(lr_up + repeat(gray(sr)-gray(up)))
// ---------------------------------------------------------------------------
__global__ void final_out(const float* __restrict__ lr_up,
                          const float* __restrict__ xcat,
                          const float* __restrict__ rr,
                          float* __restrict__ out) {
    int id = blockIdx.x * blockDim.x + threadIdx.x;
    if (id >= 131072) return;
    int b = id >> 16;
    int p = id & 65535;
    float lu[3], dp[3];
    #pragma unroll
    for (int c = 0; c < 3; ++c) {
        lu[c] = lr_up[(((size_t)b * 3 + c) << 16) + p];
        float r0 = xcat[(((size_t)b * 6 + 3 + c) << 16) + p];
        dp[c] = r0 + rr[(((size_t)b * 3 + c) << 16) + p];   // sr_pred - lr_up
    }
    float dg = 0.2989f * dp[0] + 0.587f * dp[1] + 0.114f * dp[2];
    #pragma unroll
    for (int c = 0; c < 3; ++c)
        out[(((size_t)b * 3 + c) << 16) + p] = fminf(fmaxf(lu[c] + dg, 0.0f), 1.0f);
}

// ---------------------------------------------------------------------------
// Host launcher
// ---------------------------------------------------------------------------
extern "C" void kernel_launch(void* const* d_in, const int* in_sizes, int n_in,
                              void* d_out, int out_size, void* d_ws, size_t ws_size,
                              hipStream_t stream) {
    const float* lr        = (const float*)d_in[0];
    const float* enc_w0    = (const float*)d_in[1];
    const float* enc_b0    = (const float*)d_in[2];
    const float* enc_wh    = (const float*)d_in[3];
    const float* enc_bh    = (const float*)d_in[4];
    const float* mlp_w_in  = (const float*)d_in[5];
    const float* mlp_b_in  = (const float*)d_in[6];
    const float* mlp_wh    = (const float*)d_in[7];
    const float* mlp_bh    = (const float*)d_in[8];
    const float* mlp_w_out = (const float*)d_in[9];
    const float* mlp_b_out = (const float*)d_in[10];
    const float* ref_w0    = (const float*)d_in[11];
    const float* ref_b0    = (const float*)d_in[12];
    const float* ref_w1    = (const float*)d_in[13];
    const float* ref_b1    = (const float*)d_in[14];
    const float* ref_w2    = (const float*)d_in[15];
    const float* ref_b2    = (const float*)d_in[16];

    char* base = (char*)d_ws;
    size_t off = 0;
    auto carve = [&](size_t bytes) -> char* {
        char* p = base + off;
        off = (off + bytes + 255) & ~(size_t)255;
        return p;
    };
    _Float16* wInT  = (_Float16*)carve(256 * 128 * 2);
    _Float16* whT   = (_Float16*)carve(4 * 256 * 256 * 2);
    _Float16* wOutT = (_Float16*)carve(64 * 256 * 2);
    float*    lrUp  = (float*)carve((size_t)2 * 3 * 65536 * 4);
    float*    feat0 = (float*)carve((size_t)2 * 64 * 4096 * 4);
    float*    feat1 = (float*)carve((size_t)2 * 64 * 4096 * 4);
    _Float16* X     = (_Float16*)carve((size_t)131072 * 128 * 2);
    _Float16* H0    = (_Float16*)carve((size_t)131072 * 256 * 2);
    _Float16* H1    = (_Float16*)carve((size_t)131072 * 256 * 2);
    float*    logit = (float*)carve((size_t)131072 * 64 * 4);
    float*    xcat  = (float*)carve((size_t)2 * 6 * 65536 * 4);
    float*    r0    = (float*)carve((size_t)2 * 48 * 65536 * 4);
    float*    r1    = (float*)carve((size_t)2 * 48 * 65536 * 4);
    float*    rr    = (float*)carve((size_t)2 * 3 * 65536 * 4);

    // Pack MLP weights to f16 [N][Kp]
    pack_wT<<<(256 * 128 + 255) / 256, 256, 0, stream>>>(mlp_w_in, wInT, 106, 256, 128, 256);
    for (int i = 0; i < 4; ++i)
        pack_wT<<<(256 * 256 + 255) / 256, 256, 0, stream>>>(
            mlp_wh + (size_t)i * 65536, whT + (size_t)i * 65536, 256, 256, 256, 256);
    pack_wT<<<(64 * 256 + 255) / 256, 256, 0, stream>>>(mlp_w_out, wOutT, 256, 49, 256, 64);

    // Bicubic upsample
    bicubic_up<<<(2 * 3 * 65536) / 256, 256, 0, stream>>>(lr, lrUp);

    // Encoder
    enc_conv0<<<(2 * 64 * 4096) / 256, 256, 0, stream>>>(lr, enc_w0, enc_b0, feat0);
    enc_convh<<<(2 * 64 * 4096) / 256, 256, 0, stream>>>(feat0, enc_wh + 0 * 36864, enc_bh + 0 * 64, feat1);
    enc_convh<<<(2 * 64 * 4096) / 256, 256, 0, stream>>>(feat1, enc_wh + 1 * 36864, enc_bh + 1 * 64, feat0);
    enc_convh<<<(2 * 64 * 4096) / 256, 256, 0, stream>>>(feat0, enc_wh + 2 * 36864, enc_bh + 2 * 64, feat1);

    // Build MLP input rows
    build_X<<<131072 / 256, 256, 0, stream>>>(feat1, X);

    // MLP: in -> 4 hidden -> logits (WMMA f16, f32 accum)
    // Block tile 512 rows x 64 cols; M=131072 -> 256 m-blocks.
    const int M = 131072;
    const int MB = M / 512;
    gemm_wmma<128, true, false><<<MB * (256 / 64), 256, 0, stream>>>(
        X, wInT, mlp_b_in, 256, H0, nullptr, M, 256);
    gemm_wmma<256, true, false><<<MB * (256 / 64), 256, 0, stream>>>(
        H0, whT + 0 * 65536, mlp_bh + 0 * 256, 256, H1, nullptr, M, 256);
    gemm_wmma<256, true, false><<<MB * (256 / 64), 256, 0, stream>>>(
        H1, whT + 1 * 65536, mlp_bh + 1 * 256, 256, H0, nullptr, M, 256);
    gemm_wmma<256, true, false><<<MB * (256 / 64), 256, 0, stream>>>(
        H0, whT + 2 * 65536, mlp_bh + 2 * 256, 256, H1, nullptr, M, 256);
    gemm_wmma<256, true, false><<<MB * (256 / 64), 256, 0, stream>>>(
        H1, whT + 3 * 65536, mlp_bh + 3 * 256, 256, H0, nullptr, M, 256);
    gemm_wmma<256, false, true><<<MB * (64 / 64), 256, 0, stream>>>(
        H0, wOutT, mlp_b_out, 49, nullptr, logit, M, 64);

    // Softmax + patch blend -> xcat
    softmax_patch<<<131072 / 256, 256, 0, stream>>>(logit, lr, lrUp, xcat);

    // Refinement grouped convs
    gconv3x3<<<(2 * 48 * 65536) / 256, 256, 0, stream>>>(xcat, ref_w0, ref_b0, r0, 6, 48, 1);
    gconv3x3<<<(2 * 48 * 65536) / 256, 256, 0, stream>>>(r0,   ref_w1, ref_b1, r1, 48, 48, 1);
    gconv3x3<<<(2 * 3 * 65536) / 256, 256, 0, stream>>>(r1,   ref_w2, ref_b2, rr, 48, 3, 0);

    // Final compose
    final_out<<<131072 / 256, 256, 0, stream>>>(lrUp, xcat, rr, (float*)d_out);
}
